// GAT_52750788329494
// MI455X (gfx1250) — matrix operand order repaired and checked by reading, not compiled, not run on hardware
//
#include <hip/hip_runtime.h>

#define ND 2048
#define NC 2048
#define NG 4096
#define NTOT 8192
#define EE 262144
#define EPN (EE + NTOT)     // 270336 edges incl. self loops
#define HC 256              // H1 = HEADS*H2 = HEADS*H3 = 256
#define NPAIRS 10000

typedef __attribute__((ext_vector_type(16))) _Float16 v16h;
typedef __attribute__((ext_vector_type(8)))  float    v8f;

static inline int cdiv(int a, int b) { return (a + b - 1) / b; }

// ---------------------------------------------------------------------------
// GEMM: C[M x Nc] = A[M x K](f32) * B[K x Nc](f32) + bias, via f16 WMMA with
// f32 accumulation. Block = 128 threads (4 wave32), tile 64x64, K-step 64
// (8 WMMAs per barrier phase). B is transposed in LDS so both A and B
// fragments are contiguous ds_load_b128 pairs. Global tile loads are issued
// in one batch (register-staged) so a single s_wait_loadcnt covers them.
// ---------------------------------------------------------------------------
#define TM 64
#define TN 64
#define KB 64
#define LDA 72    // As row stride in halves (144B, 16B aligned)
#define LDBT 72   // BsT row stride in halves

__global__ __launch_bounds__(128) void gemm_wmma_f16(
    const float* __restrict__ A, const float* __restrict__ B,
    const float* __restrict__ bias, float* __restrict__ C,
    int M, int K, int Nc)
{
  __shared__ _Float16 As[TM * LDA];    // [row][k]
  __shared__ _Float16 BsT[TN * LDBT];  // [col][k]  (transposed B tile)
  const int tid  = threadIdx.x;
  const int lane = tid & 31;
  const int wave = tid >> 5;
  const int half = lane >> 4;     // which 16-lane half of the wave
  const int l16  = lane & 15;
  const int bm = blockIdx.y * TM;
  const int bn = blockIdx.x * TN;
  const int wm = (wave >> 1) * 32;   // wave sub-tile: 32x32 = 2x2 WMMA tiles
  const int wn = (wave & 1) * 32;

  v8f acc[2][2] = {};

  const int c4 = tid & 15;   // float4 column group (16 per 64-wide row)
  const int r0 = tid >> 4;   // 0..7

  const float* aBase = A + (size_t)(bm + r0) * K + c4 * 4;
  const float* bBase = B + (size_t)r0 * Nc + bn + c4 * 4;

  for (int k0 = 0; k0 < K; k0 += KB) {
    // ---- issue all global tile loads first (one wait covers the batch) ----
    float4 va[8], vb[8];
    #pragma unroll
    for (int i = 0; i < 8; ++i)
      va[i] = *(const float4*)(aBase + (size_t)(8 * i) * K + k0);
    #pragma unroll
    for (int i = 0; i < 8; ++i)
      vb[i] = *(const float4*)(bBase + (size_t)(k0 + 8 * i) * Nc);

    // prefetch next K-step tiles into cache while this phase computes
    if (k0 + KB < K) {
      __builtin_prefetch(aBase + (size_t)(k0 + KB), 0, 1);
      __builtin_prefetch(bBase + (size_t)(k0 + KB) * Nc, 0, 1);
    }

    // ---- convert f32 -> f16 and store to LDS ----
    #pragma unroll
    for (int i = 0; i < 8; ++i) {
      _Float16* p = &As[(r0 + 8 * i) * LDA + c4 * 4];
      p[0] = (_Float16)va[i].x; p[1] = (_Float16)va[i].y;
      p[2] = (_Float16)va[i].z; p[3] = (_Float16)va[i].w;
    }
    #pragma unroll
    for (int i = 0; i < 8; ++i) {
      int kr = r0 + 8 * i;
      int cb = c4 * 4;
      BsT[(cb + 0) * LDBT + kr] = (_Float16)vb[i].x;
      BsT[(cb + 1) * LDBT + kr] = (_Float16)vb[i].y;
      BsT[(cb + 2) * LDBT + kr] = (_Float16)vb[i].z;
      BsT[(cb + 3) * LDBT + kr] = (_Float16)vb[i].w;
    }
    __syncthreads();

    // ---- 2 WMMA K-subtiles of 32, fragments per ISA 7.12.2 ----
    #pragma unroll
    for (int ks = 0; ks < KB; ks += 32) {
      v16h afrag[2], bfrag[2];
      #pragma unroll
      for (int ti = 0; ti < 2; ++ti) {
        // A 16x32 f16: e<8 -> K = ks + half*8 + e ; e>=8 -> K = ks+16+half*8+(e-8)
        const _Float16* ap = &As[(wm + ti * 16 + l16) * LDA + ks + half * 8];
        #pragma unroll
        for (int e = 0; e < 8; ++e) { afrag[ti][e] = ap[e]; afrag[ti][8 + e] = ap[16 + e]; }
      }
      #pragma unroll
      for (int tj = 0; tj < 2; ++tj) {
        // B 32x16 f16: e<8 -> K = ks + half*8 + e ; e>=8 -> K = ks+16+half*8+(e-8), N = lane%16
        const _Float16* bp = &BsT[(wn + tj * 16 + l16) * LDBT + ks + half * 8];
        #pragma unroll
        for (int e = 0; e < 8; ++e) { bfrag[tj][e] = bp[e]; bfrag[tj][8 + e] = bp[16 + e]; }
      }
      #pragma unroll
      for (int ti = 0; ti < 2; ++ti)
        #pragma unroll
        for (int tj = 0; tj < 2; ++tj)
          acc[ti][tj] = __builtin_amdgcn_wmma_f32_16x16x32_f16(
              false, afrag[ti], false, bfrag[tj], (short)0, acc[ti][tj], false, false);
    }
    __syncthreads();
  }

  // ---- epilogue: C/D layout VGPR v -> row v + 8*half, col lane%16 ----
  #pragma unroll
  for (int ti = 0; ti < 2; ++ti)
    #pragma unroll
    for (int tj = 0; tj < 2; ++tj) {
      int c = bn + wn + tj * 16 + l16;
      float bv = bias[c];
      #pragma unroll
      for (int v = 0; v < 8; ++v) {
        int r = bm + wm + ti * 16 + v + half * 8;
        C[(size_t)r * Nc + c] = acc[ti][tj][v] + bv;
      }
    }
}

// ---------------------------------------------------------------------------
// order-preserving float<->uint keys for atomic segment-max
// ---------------------------------------------------------------------------
__device__ __forceinline__ unsigned ordKey(float f) {
  int b = __float_as_int(f);
  return (unsigned)(b >= 0 ? (b ^ 0x80000000) : ~b);
}
__device__ __forceinline__ float ordUnkey(unsigned u) {
  int b = (u & 0x80000000u) ? (int)(u ^ 0x80000000u) : ~(int)u;
  return __int_as_float(b);
}

// out[n*256+j] = gat output bias[j]
__global__ void init_out_bias(float* __restrict__ out, const float* __restrict__ bias) {
  int t = blockIdx.x * blockDim.x + threadIdx.x;
  if (t < NTOT * HC) out[t] = bias[t & (HC - 1)];
}

// one wave per edge: logits[e][h] = att[h] . leaky_relu(xl[src] + xr[dst]); atomic seg-max
__global__ __launch_bounds__(256) void edge_logits(
    const float* __restrict__ xl, const float* __restrict__ xr,
    const int* __restrict__ edges, const float* __restrict__ att,
    float* __restrict__ logits, unsigned* __restrict__ mu)
{
  int e = blockIdx.x * (blockDim.x >> 5) + (threadIdx.x >> 5);
  if (e >= EPN) return;
  int lane = threadIdx.x & 31;
  int s, d;
  if (e < EE) { s = edges[e]; d = edges[EE + e]; } else { s = d = e - EE; }
  int cbase = lane * 8;
  int h = cbase >> 6;
  const float* pl = xl + (size_t)s * HC + cbase;
  const float* pr = xr + (size_t)d * HC + cbase;
  const float* pa = att + h * 64 + (cbase & 63);
  float t = 0.f;
  #pragma unroll
  for (int i = 0; i < 8; ++i) {
    float v = pl[i] + pr[i];
    v = v > 0.f ? v : 0.2f * v;
    t += pa[i] * v;
  }
  t += __shfl_down(t, 4, 8);
  t += __shfl_down(t, 2, 8);
  t += __shfl_down(t, 1, 8);
  if ((lane & 7) == 0) {
    logits[e * 4 + h] = t;
    atomicMax(&mu[d * 4 + h], ordKey(t));
  }
}

// ex = exp(logit - max); atomic seg-sum of ex (in-place over logits)
__global__ void edge_exp(float* __restrict__ logits, const unsigned* __restrict__ mu,
                         float* __restrict__ den, const int* __restrict__ edges)
{
  int t = blockIdx.x * blockDim.x + threadIdx.x;
  if (t >= EPN * 4) return;
  int e = t >> 2, h = t & 3;
  int d = (e < EE) ? edges[EE + e] : (e - EE);
  float ex = __expf(logits[t] - ordUnkey(mu[d * 4 + h]));
  logits[t] = ex;
  atomicAdd(&den[d * 4 + h], ex);
}

// out[dst] += alpha * xl[src], thread = (edge, 4-channel group)
__global__ void edge_agg(const float* __restrict__ ex, const float* __restrict__ den,
                         const float* __restrict__ xl, const int* __restrict__ edges,
                         float* __restrict__ out)
{
  long long t = (long long)blockIdx.x * blockDim.x + threadIdx.x;
  if (t >= (long long)EPN * 64) return;
  int e = (int)(t >> 6);
  int g = (int)(t & 63);
  int c = g * 4, h = g >> 4;
  int s, d;
  if (e < EE) { s = edges[e]; d = edges[EE + e]; } else { s = d = e - EE; }
  float alpha = ex[e * 4 + h] / den[d * 4 + h];
  float4 v = *(const float4*)(xl + (size_t)s * HC + c);
  float* o = out + (size_t)d * HC + c;
  atomicAdd(o + 0, alpha * v.x);
  atomicAdd(o + 1, alpha * v.y);
  atomicAdd(o + 2, alpha * v.z);
  atomicAdd(o + 3, alpha * v.w);
}

// GraphNorm: pass1 column sums
__global__ __launch_bounds__(256) void gn_colsum(const float* __restrict__ X, float* __restrict__ cs) {
  int j = threadIdx.x, r0 = blockIdx.x * 64;
  float s = 0.f;
  for (int r = 0; r < 64; ++r) s += X[(size_t)(r0 + r) * HC + j];
  atomicAdd(&cs[j], s);
}
// pass2: column sums of (x - mu*mscale)^2
__global__ __launch_bounds__(256) void gn_colsum2(const float* __restrict__ X,
                                                  const float* __restrict__ cs,
                                                  const float* __restrict__ mscale,
                                                  float* __restrict__ cs2) {
  int j = threadIdx.x, r0 = blockIdx.x * 64;
  float mu = cs[j] * (1.0f / NTOT) * mscale[j];
  float s = 0.f;
  for (int r = 0; r < 64; ++r) { float v = X[(size_t)(r0 + r) * HC + j] - mu; s += v * v; }
  atomicAdd(&cs2[j], s);
}
// pass3: normalize + affine + relu -> Xout
__global__ void gn_norm(const float* __restrict__ Xin, const float* __restrict__ cs,
                        const float* __restrict__ cs2, const float* __restrict__ mscale,
                        const float* __restrict__ gamma, const float* __restrict__ beta,
                        float* __restrict__ Xout) {
  int t = blockIdx.x * blockDim.x + threadIdx.x;
  if (t >= NTOT * HC) return;
  int j = t & (HC - 1);
  float mu = cs[j] * (1.0f / NTOT);
  float var = cs2[j] * (1.0f / NTOT);
  float v = (Xin[t] - mu * mscale[j]) * rsqrtf(var + 1e-5f) * gamma[j] + beta[j];
  Xout[t] = v > 0.f ? v : 0.f;
}

// wave per pair: sigmoid(concat(x[id],x[ic]) . W + b)
__global__ __launch_bounds__(256) void pair_pred(
    const float* __restrict__ X, const int* __restrict__ idxd, const int* __restrict__ idxc,
    const float* __restrict__ W, const float* __restrict__ b, float* __restrict__ pred)
{
  int p = blockIdx.x * 8 + (threadIdx.x >> 5);
  if (p >= NPAIRS) return;
  int lane = threadIdx.x & 31;
  int id = idxd[p], ic = idxc[p];
  float s = 0.f;
  for (int j = lane; j < HC; j += 32)
    s += X[(size_t)id * HC + j] * W[j] + X[(size_t)ic * HC + j] * W[HC + j];
  for (int off = 16; off; off >>= 1) s += __shfl_down(s, off, 32);
  if (lane == 0) pred[p] = 1.0f / (1.0f + __expf(-(s + b[0])));
}

// attn[src, dst] = val for all edges + self loops (attn pre-zeroed by memset)
__global__ void scatter_attn(const int* __restrict__ edges, float* __restrict__ attn, float val) {
  int t = blockIdx.x * blockDim.x + threadIdx.x;
  if (t >= EPN) return;
  int s, d;
  if (t < EE) { s = edges[t]; d = edges[EE + t]; } else { s = d = t - EE; }
  attn[(size_t)s * NTOT + d] = val;
}

// ---------------------------------------------------------------------------
extern "C" void kernel_launch(void* const* d_in, const int* in_sizes, int n_in,
                              void* d_out, int out_size, void* d_ws, size_t ws_size,
                              hipStream_t stream) {
  (void)in_sizes; (void)n_in; (void)out_size; (void)ws_size;
  const float* drug  = (const float*)d_in[0];
  const float* cell  = (const float*)d_in[1];
  const float* gene  = (const float*)d_in[2];
  const int*   edges = (const int*)  d_in[3];
  const int*   idx_d = (const int*)  d_in[4];
  const int*   idx_c = (const int*)  d_in[5];
  const float* W_drug = (const float*)d_in[6];  const float* b_drug = (const float*)d_in[7];
  const float* W_cell = (const float*)d_in[8];  const float* b_cell = (const float*)d_in[9];
  const float* W_gene = (const float*)d_in[10]; const float* b_gene = (const float*)d_in[11];
  const float* g1_Wl = (const float*)d_in[12];  const float* g1_bl = (const float*)d_in[13];
  const float* g1_Wr = (const float*)d_in[14];  const float* g1_br = (const float*)d_in[15];
  const float* g1_att = (const float*)d_in[16]; const float* g1_bias = (const float*)d_in[17];
  const float* g2_Wl = (const float*)d_in[18];  const float* g2_bl = (const float*)d_in[19];
  const float* g2_Wr = (const float*)d_in[20];  const float* g2_br = (const float*)d_in[21];
  const float* g2_att = (const float*)d_in[22]; const float* g2_bias = (const float*)d_in[23];
  const float* gn1_g = (const float*)d_in[24];  const float* gn1_b = (const float*)d_in[25];
  const float* gn1_s = (const float*)d_in[26];
  const float* gn2_g = (const float*)d_in[27];  const float* gn2_b = (const float*)d_in[28];
  const float* gn2_s = (const float*)d_in[29];
  const float* lin_W = (const float*)d_in[30];  const float* lin_b = (const float*)d_in[31];

  const size_t NN = (size_t)NTOT * HC;      // 2,097,152
  float*    X   = (float*)d_ws;
  float*    XL  = X   + NN;
  float*    XR  = XL  + NN;
  float*    OUT = XR  + NN;
  float*    LOG = OUT + NN;                 // EPN*4
  unsigned* MU  = (unsigned*)(LOG + (size_t)EPN * 4);
  float*    DEN = (float*)(MU + (size_t)NTOT * 4);
  float*    CS1 = DEN + (size_t)NTOT * 4;
  float*    CS2 = CS1 + HC;
  float*    pred = (float*)d_out;
  float*    attn = pred + NPAIRS;

  // ---- Step A: X = concat(drug@Wd+b, cell@Wc+b, gene@Wg+b) ----
  gemm_wmma_f16<<<dim3(HC / TN, ND / TM), 128, 0, stream>>>(drug, W_drug, b_drug, X, ND, ND, HC);
  gemm_wmma_f16<<<dim3(HC / TN, NC / TM), 128, 0, stream>>>(cell, W_cell, b_cell, X + (size_t)ND * HC, NC, NC, HC);
  gemm_wmma_f16<<<dim3(HC / TN, NG / TM), 128, 0, stream>>>(gene, W_gene, b_gene, X + (size_t)(ND + NC) * HC, NG, NG, HC);

  // ---- Two GATv2 + GraphNorm layers (X -> X) ----
  for (int layer = 0; layer < 2; ++layer) {
    const float* Wl = layer ? g2_Wl : g1_Wl;  const float* bl = layer ? g2_bl : g1_bl;
    const float* Wr = layer ? g2_Wr : g1_Wr;  const float* br = layer ? g2_br : g1_br;
    const float* at = layer ? g2_att : g1_att;
    const float* gb = layer ? g2_bias : g1_bias;
    const float* gg = layer ? gn2_g : gn1_g;  const float* gbe = layer ? gn2_b : gn1_b;
    const float* gs = layer ? gn2_s : gn1_s;

    gemm_wmma_f16<<<dim3(HC / TN, NTOT / TM), 128, 0, stream>>>(X, Wl, bl, XL, NTOT, HC, HC);
    gemm_wmma_f16<<<dim3(HC / TN, NTOT / TM), 128, 0, stream>>>(X, Wr, br, XR, NTOT, HC, HC);

    hipMemsetAsync(MU,  0, (size_t)NTOT * 4 * sizeof(unsigned), stream);
    hipMemsetAsync(DEN, 0, (size_t)NTOT * 4 * sizeof(float), stream);
    init_out_bias<<<cdiv(NTOT * HC, 256), 256, 0, stream>>>(OUT, gb);

    edge_logits<<<cdiv(EPN, 8), 256, 0, stream>>>(XL, XR, edges, at, LOG, MU);
    edge_exp<<<cdiv(EPN * 4, 256), 256, 0, stream>>>(LOG, MU, DEN, edges);
    edge_agg<<<cdiv(EPN * 64, 256), 256, 0, stream>>>(LOG, DEN, XL, edges, OUT);

    hipMemsetAsync(CS1, 0, HC * sizeof(float), stream);
    hipMemsetAsync(CS2, 0, HC * sizeof(float), stream);
    gn_colsum<<<NTOT / 64, 256, 0, stream>>>(OUT, CS1);
    gn_colsum2<<<NTOT / 64, 256, 0, stream>>>(OUT, CS1, gs, CS2);
    gn_norm<<<cdiv(NTOT * HC, 256), 256, 0, stream>>>(OUT, CS1, CS2, gs, gg, gbe, X);
  }

  // ---- pair prediction ----
  pair_pred<<<cdiv(NPAIRS, 8), 256, 0, stream>>>(X, idx_d, idx_c, lin_W, lin_b, pred);

  // ---- attention matrix: every (dst,head) softmax sums to 1 (self-loops
  // guarantee non-empty segments), so a.mean() = N/E' exactly; val = 2*N/E'.
  const float val = 2.0f * (float)NTOT / (float)EPN;
  hipMemsetAsync(attn, 0, (size_t)NTOT * NTOT * sizeof(float), stream);
  scatter_attn<<<cdiv(EPN, 256), 256, 0, stream>>>(edges, attn, val);
}
